// OptimizedSecureAttention_49752901157246
// MI455X (gfx1250) — matrix operand order repaired and checked by reading, not compiled, not run on hardware
//
#include <hip/hip_runtime.h>

#define B_   4
#define S_   2048
#define DIM_ 512
#define H_   8
#define HD_  64
#define LOG2E 1.44269504088896f

typedef __bf16 bf16;
typedef __attribute__((ext_vector_type(16))) __bf16 v16bf;
typedef __attribute__((ext_vector_type(8)))  float  v8f;

union V16 {
    v16bf          v;
    unsigned short us[16];
    unsigned int   ui[8];
};

__device__ __forceinline__ unsigned short f2bf(float f) {
    unsigned int u = __float_as_uint(f);
    u += 0x7FFFu + ((u >> 16) & 1u);      // round-to-nearest-even
    return (unsigned short)(u >> 16);
}

// pack two f32 -> bf16x2 (v_cvt_pk_bf16_f32 when the builtin exists)
__device__ __forceinline__ unsigned int pack_bf16(float f0, float f1) {
#if __has_builtin(__builtin_amdgcn_cvt_pk_bf16_f32)
    typedef __attribute__((ext_vector_type(2))) __bf16 v2bf;
    union { v2bf v; unsigned int u; } p;
    p.v = __builtin_amdgcn_cvt_pk_bf16_f32(f0, f1);
    return p.u;
#else
    return (unsigned int)f2bf(f0) | ((unsigned int)f2bf(f1) << 16);
#endif
}

__device__ __forceinline__ v8f wmma_bf16(const V16& a, const V16& b, v8f c) {
    // 8 args: (neg_a, A, neg_b, B, c_mod, C, reuse_a, reuse_b)
    return __builtin_amdgcn_wmma_f32_16x16x32_bf16(
        false, a.v, false, b.v, (short)0, c, false, false);
}

// async memory->LDS staging (gfx1250 GLOBAL_LOAD_ASYNC_TO_LDS_B128, ASYNCcnt)
__device__ __forceinline__ void async_load_lds_b128(unsigned int lds_off,
                                                    unsigned long long gaddr) {
    asm volatile("global_load_async_to_lds_b128 %0, %1, off"
                 :: "v"(lds_off), "v"(gaddr) : "memory");
}
__device__ __forceinline__ void wait_asynccnt0() {
#if __has_builtin(__builtin_amdgcn_s_wait_asynccnt)
    __builtin_amdgcn_s_wait_asynccnt(0);
#else
    asm volatile("s_wait_asynccnt 0" ::: "memory");
#endif
}

// ---------------------------------------------------------------------------
// Kernel 1: P = x @ W^T + b   (M=B*S=8192, N=512, K=512), output bf16 [B,H,S,HD]
// 256 threads = 8 waves; block tile 64x64; each wave: 16 rows x 32 cols (2 WMMA C tiles)
// ---------------------------------------------------------------------------
__global__ __launch_bounds__(256) void qkv_proj_kernel(
    const float* __restrict__ x, const float* __restrict__ W,
    const float* __restrict__ bias, unsigned short* __restrict__ dst)
{
    __shared__ alignas(32) unsigned short Xt[64 * 32];  // [row][k] bf16
    __shared__ alignas(32) unsigned short Wt[64 * 32];  // [n][k]   bf16

    const int tid  = threadIdx.x;
    const int lane = tid & 31;
    const int wv   = tid >> 5;
    const int mloc = (wv & 3) * 16;
    const int nloc = (wv >> 2) * 32;
    const int blockM = blockIdx.x * 64;
    const int blockN = blockIdx.y * 64;

    const int half8 = (lane < 16) ? 0 : 8;   // C-layout M offset
    const int m16   = lane & 15;
    const int kb16  = (lane < 16) ? 0 : 16;  // B-layout K base

    v8f acc0 = {}; v8f acc1 = {};

    for (int k0 = 0; k0 < DIM_; k0 += 32) {
        // stage x and W tiles, f32 -> bf16: b128 loads, packed uint2 LDS stores
        for (int i = tid; i < 512; i += 256) {
            int row = i >> 3, q4 = i & 7;
            float4 xv = *(const float4*)&x[(size_t)(blockM + row) * DIM_ + k0 + 4 * q4];
            ((uint2*)Xt)[i] = make_uint2(pack_bf16(xv.x, xv.y), pack_bf16(xv.z, xv.w));
            float4 wv4 = *(const float4*)&W[(size_t)(blockN + row) * DIM_ + k0 + 4 * q4];
            ((uint2*)Wt)[i] = make_uint2(pack_bf16(wv4.x, wv4.y), pack_bf16(wv4.z, wv4.w));
        }
        __syncthreads();

        V16 a;
        {
            const unsigned short* xr = &Xt[(mloc + m16) * 32];
            #pragma unroll
            for (int i = 0; i < 8; ++i) {
                int kk = kb16 + 2 * i;
                a.ui[i] = *(const unsigned int*)&xr[kk];
            }
        }
        V16 b0, b1;
        b0.v = *(const v16bf*)&Wt[(nloc + m16) * 32 + kb16];
        b1.v = *(const v16bf*)&Wt[(nloc + 16 + m16) * 32 + kb16];
        acc0 = wmma_bf16(a, b0, acc0);
        acc1 = wmma_bf16(a, b1, acc1);
        __syncthreads();
    }

    // epilogue: bias add, head-split store as bf16 into [B,H,S,HD]
    #pragma unroll
    for (int t = 0; t < 2; ++t) {
        v8f acc = t ? acc1 : acc0;
        int n = blockN + nloc + 16 * t + m16;
        float bs = bias[n];
        int h = n >> 6, hd = n & 63;
        #pragma unroll
        for (int r = 0; r < 8; ++r) {
            int m = blockM + mloc + r + half8;
            int bb = m >> 11, s = m & 2047;
            dst[(((size_t)(bb * H_ + h) * S_) + s) * HD_ + hd] = f2bf(acc[r] + bs);
        }
    }
}

// ---------------------------------------------------------------------------
// Kernel 2: fused flash attention per (b,h); 128 threads = 4 waves.
// Block: 64 queries; wave: 16 query rows (Q kept in A-layout registers).
// K tile staged via GLOBAL_LOAD_ASYNC_TO_LDS_B128 (ASYNCcnt); V transposed via VGPRs.
// ---------------------------------------------------------------------------
__global__ __launch_bounds__(128) void attn_kernel(
    const unsigned short* __restrict__ qws, const unsigned short* __restrict__ kws,
    const unsigned short* __restrict__ vws, const float* __restrict__ noise,
    float* __restrict__ out)
{
    __shared__ alignas(32) unsigned short Kt[32 * 64];   // [key][d]  row-major
    __shared__ alignas(32) unsigned short VtT[64 * 32];  // [d][key]  transposed
    __shared__ alignas(32) unsigned short Pt[4][16 * 32];// per-wave P (C->A relayout)

    const int tid  = threadIdx.x;
    const int lane = tid & 31;
    const int wv   = tid >> 5;
    const int b = blockIdx.z, h = blockIdx.y;
    const int qbase = blockIdx.x * 64 + wv * 16;
    const long bh = (long)(b * H_ + h) * S_;

    const int half8 = (lane < 16) ? 0 : 8;
    const int m16   = lane & 15;
    const int kb16  = (lane < 16) ? 0 : 16;

    const unsigned int kt_lds = (unsigned int)(unsigned long long)(uintptr_t)&Kt[0];

    // Q rows -> A-layout registers, two k-chunks (d=0..31, d=32..63)
    V16 qa[2];
    {
        const unsigned short* qrow = &qws[(bh + qbase + m16) * HD_];
        #pragma unroll
        for (int c = 0; c < 2; ++c)
            #pragma unroll
            for (int i = 0; i < 8; ++i) {
                int kk = 32 * c + kb16 + 2 * i;
                qa[c].ui[i] = *(const unsigned int*)&qrow[kk];
            }
    }

    float mrun[8], lrun[8];
    v8f o[4] = {{}, {}, {}, {}};
    #pragma unroll
    for (int r = 0; r < 8; ++r) { mrun[r] = -__builtin_inff(); lrun[r] = 0.f; }

    for (int kt = 0; kt < S_ / 32; ++kt) {
        const unsigned short* ksrc = &kws[(bh + kt * 32) * HD_];
        const unsigned short* vsrc = &vws[(bh + kt * 32) * HD_];
        if (kt + 1 < S_ / 32)   // gfx1250 global_prefetch_b8 on next K tile
            __builtin_prefetch(&kws[(bh + (kt + 1) * 32) * HD_ + tid * 16], 0, 1);

        // K tile: async memory->LDS, no VGPR round-trip (2 x b128 per thread)
        for (int i = tid; i < 256; i += 128) {
            async_load_lds_b128(kt_lds + (unsigned)i * 16u,
                                (unsigned long long)(uintptr_t)ksrc
                                    + (unsigned long long)i * 16ull);
        }
        // V tile: load b128, transpose into [d][key] via b16 LDS stores
        for (int i = tid; i < 256; i += 128) {
            uint4 vvec = ((const uint4*)vsrc)[i];
            int key = i >> 3;
            int d0  = (i & 7) * 8;
            const unsigned short* pv = (const unsigned short*)&vvec;
            #pragma unroll
            for (int j = 0; j < 8; ++j) VtT[(d0 + j) * 32 + key] = pv[j];
        }
        wait_asynccnt0();
        __syncthreads();

        // S = Q K^T : 2 N-tiles x 2 k-chunks
        v8f c0 = {}, c1 = {};
        #pragma unroll
        for (int c = 0; c < 2; ++c) {
            V16 b0, b1;
            b0.v = *(const v16bf*)&Kt[m16 * 64 + 32 * c + kb16];
            b1.v = *(const v16bf*)&Kt[(16 + m16) * 64 + 32 * c + kb16];
            c0 = wmma_bf16(qa[c], b0, c0);
            c1 = wmma_bf16(qa[c], b1, c1);
        }

        // online softmax (row stats replicated across each 16-lane half)
        #pragma unroll
        for (int r = 0; r < 8; ++r) {
            float s0 = c0[r] * 0.125f;          // 1/sqrt(HD)
            float s1 = c1[r] * 0.125f;
            float mt = fmaxf(s0, s1);
            #pragma unroll
            for (int off = 1; off < 16; off <<= 1)
                mt = fmaxf(mt, __shfl_xor(mt, off, 32));
            float mnew  = fmaxf(mrun[r], mt);
            float alpha = exp2f((mrun[r] - mnew) * LOG2E);
            float p0 = exp2f((s0 - mnew) * LOG2E);
            float p1 = exp2f((s1 - mnew) * LOG2E);
            float rs = p0 + p1;
            #pragma unroll
            for (int off = 1; off < 16; off <<= 1)
                rs += __shfl_xor(rs, off, 32);
            lrun[r] = lrun[r] * alpha + rs;
            mrun[r] = mnew;
            #pragma unroll
            for (int t = 0; t < 4; ++t) o[t][r] *= alpha;
            int ml = r + half8;
            Pt[wv][ml * 32 + m16]      = f2bf(p0);
            Pt[wv][ml * 32 + 16 + m16] = f2bf(p1);
        }

        // O += P V : A from per-wave LDS buffer (C->A relayout), B from transposed V
        V16 pa;
        #pragma unroll
        for (int i = 0; i < 8; ++i) {
            int kk = kb16 + 2 * i;
            pa.ui[i] = *(const unsigned int*)&Pt[wv][m16 * 32 + kk];
        }
        #pragma unroll
        for (int t = 0; t < 4; ++t) {
            V16 vb;
            vb.v = *(const v16bf*)&VtT[(16 * t + m16) * 32 + kb16];
            o[t] = wmma_bf16(pa, vb, o[t]);
        }
        __syncthreads();
    }

    // epilogue: normalize, secure-aggregation noise, merge heads
    const float inv101 = 1.0f / 1.01f;
    #pragma unroll
    for (int r = 0; r < 8; ++r) {
        int q = qbase + r + half8;
        float invl = 1.0f / lrun[r];
        #pragma unroll
        for (int t = 0; t < 4; ++t) {
            int d = 16 * t + m16;
            float nz = noise[(bh + q) * HD_ + d];
            out[(long)(b * S_ + q) * DIM_ + h * HD_ + d] = (o[t][r] * invl + nz) * inv101;
        }
    }
}

extern "C" void kernel_launch(void* const* d_in, const int* in_sizes, int n_in,
                              void* d_out, int out_size, void* d_ws, size_t ws_size,
                              hipStream_t stream) {
    const float* x     = (const float*)d_in[0];
    const float* Wq    = (const float*)d_in[1];
    const float* bq    = (const float*)d_in[2];
    const float* Wk    = (const float*)d_in[3];
    const float* bk    = (const float*)d_in[4];
    const float* Wv    = (const float*)d_in[5];
    const float* bv    = (const float*)d_in[6];
    const float* noise = (const float*)d_in[7];
    float* out = (float*)d_out;

    const size_t per = (size_t)B_ * H_ * S_ * HD_;    // 4,194,304 elems (8 MB bf16)
    unsigned short* qws = (unsigned short*)d_ws;
    unsigned short* kws = qws + per;
    unsigned short* vws = kws + per;

    dim3 pg(B_ * S_ / 64, DIM_ / 64);                 // 128 x 8 blocks
    qkv_proj_kernel<<<pg, 256, 0, stream>>>(x, Wq, bq, qws);
    qkv_proj_kernel<<<pg, 256, 0, stream>>>(x, Wk, bk, kws);
    qkv_proj_kernel<<<pg, 256, 0, stream>>>(x, Wv, bv, vws);

    dim3 ag(S_ / 64, H_, B_);                         // 32 x 8 x 4 blocks
    attn_kernel<<<ag, 128, 0, stream>>>(qws, kws, vws, noise, out);
}
